// GPT_20650202759734
// MI455X (gfx1250) — matrix operand order repaired and checked by reading, not compiled, not run on hardware
//
#include <hip/hip_runtime.h>

// ---------------------------------------------------------------------------
// MI455X (gfx1250) GPT forward. All matmuls via v_wmma_f32_16x16x32_bf16.
// Compute-bound (~400 GFLOP vs ~14us of HBM traffic at 23.3 TB/s), so every
// GEMM (QKV/proj/MLP/head) and both attention matmuls run on the WMMA path
// with bf16 inputs / f32 accumulation. Wave32, 16x16x32 tiles.
// Wave-uniform coordinates are forced into SGPRs via readfirstlane so the
// backend emits uniform s_cbranch control flow (EXEC stays all-1s around
// WMMA, per ISA requirement) and scalar tile-base addressing.
// ---------------------------------------------------------------------------

typedef __bf16 bf16;
typedef __attribute__((ext_vector_type(16))) __bf16 v16bf;
typedef __attribute__((ext_vector_type(8)))  __bf16 v8bf;
typedef __attribute__((ext_vector_type(8)))  float  v8f;

#define NEGINF (-__builtin_inff())

static __device__ __forceinline__ int uni(int v) {
  return __builtin_amdgcn_readfirstlane(v);   // assert wave-uniform -> SGPR
}

static __device__ __forceinline__ v8f vzero8() {
  v8f z;
#pragma unroll
  for (int i = 0; i < 8; ++i) z[i] = 0.0f;
  return z;
}

static __device__ __forceinline__ v8f wmma_bf16(v16bf a, v16bf b, v8f c) {
  // D = A(16x32 bf16) * B(32x16 bf16) + C(16x16 f32)
  return __builtin_amdgcn_wmma_f32_16x16x32_bf16(false, a, false, b,
                                                 (short)0, c, false, false);
}

// A-fragment (16x32, row-major source, leading dim ld in elements):
// lane m (0..15): row m, K chunks {0..7} and {16..23}
// lane m+16     : row m, K chunks {8..15} and {24..31}
static __device__ __forceinline__ v16bf load_a_frag(const bf16* __restrict__ base,
                                                    int ld, int lane) {
  const int m  = lane & 15;
  const int kh = lane >> 4;
  const bf16* p = base + (size_t)m * ld + kh * 8;
  v8bf lo = *(const v8bf*)(p);
  v8bf hi = *(const v8bf*)(p + 16);
  v16bf r;
#pragma unroll
  for (int i = 0; i < 8; ++i) { r[i] = lo[i]; r[i + 8] = hi[i]; }
  return r;
}

// B-fragment (32x16). Source is W[n, k] row-major with leading dim ld, so that
// B[k][n] = W[n][k] (i.e. the GEMM computes A @ W^T).
// lane n (0..15): column n, K {0..7} and {16..23} ; lane n+16: column n, K {8..15} and {24..31}.
static __device__ __forceinline__ v16bf load_b_frag(const bf16* __restrict__ base,
                                                    int ld, int lane) {
  const int n  = lane & 15;
  const int kh = lane >> 4;
  const bf16* p = base + (size_t)n * ld + kh * 8;
  v8bf lo = *(const v8bf*)(p);
  v8bf hi = *(const v8bf*)(p + 16);
  v16bf r;
#pragma unroll
  for (int i = 0; i < 8; ++i) { r[i] = lo[i]; r[i + 8] = hi[i]; }
  return r;
}

static __device__ __forceinline__ float lane_bcast(float v, int srcLane) {
  return __builtin_bit_cast(float,
      __builtin_amdgcn_ds_bpermute(srcLane << 2, __builtin_bit_cast(int, v)));
}
static __device__ __forceinline__ float lane_xor16(float v, int lane) {
  return lane_bcast(v, lane ^ 16);
}

// ---------------------------------------------------------------------------
// Elementwise fp32 -> bf16 weight conversion (staged per-matrix before use).
// ---------------------------------------------------------------------------
__global__ void __launch_bounds__(256) cvt_kernel(const float* __restrict__ in,
                                                  bf16* __restrict__ out,
                                                  long n) {
  long i = (long)blockIdx.x * blockDim.x + threadIdx.x;
  const long stride = (long)gridDim.x * blockDim.x;
  for (; i < n; i += stride) out[i] = (bf16)in[i];
}

// ---------------------------------------------------------------------------
// Embedding gather: x[b,t,:] = tok[i0] + pos[i1] + ab[i2]
// ---------------------------------------------------------------------------
__global__ void __launch_bounds__(256) embed_kernel(
    const int* __restrict__ idx, const float* __restrict__ tok,
    const float* __restrict__ pos, const float* __restrict__ ab,
    float* __restrict__ x, int Tn, int Cn) {
  const int bt = blockIdx.x;
  const int b = bt / Tn;
  const int t = bt - b * Tn;
  const int* ib = idx + (size_t)b * 3 * Tn;
  const int i0 = ib[t], i1 = ib[Tn + t], i2 = ib[2 * Tn + t];
  for (int c = threadIdx.x; c < Cn; c += 256) {
    x[(size_t)bt * Cn + c] = tok[(size_t)i0 * Cn + c] +
                             pos[(size_t)i1 * Cn + c] +
                             ab[(size_t)i2 * Cn + c];
  }
}

// ---------------------------------------------------------------------------
// LayerNorm (one block per row of C=1024), fp32 in -> bf16 out.
// ---------------------------------------------------------------------------
__global__ void __launch_bounds__(256) ln_kernel(
    const float* __restrict__ x, const float* __restrict__ w,
    const float* __restrict__ b, bf16* __restrict__ out, int Cn) {
  const int row = blockIdx.x;
  const int tid = threadIdx.x;
  const float* xr = x + (size_t)row * Cn;
  float v4[4];
  float s = 0.0f, ss = 0.0f;
#pragma unroll
  for (int j = 0; j < 4; ++j) {
    float t = xr[tid + j * 256];
    v4[j] = t; s += t; ss += t * t;
  }
  __shared__ float sh[512];
  sh[tid] = s; sh[tid + 256] = ss;
  __syncthreads();
  for (int off = 128; off > 0; off >>= 1) {
    if (tid < off) {
      sh[tid] += sh[tid + off];
      sh[tid + 256] += sh[tid + 256 + off];
    }
    __syncthreads();
  }
  const float mean = sh[0] / Cn;
  const float var  = sh[256] / Cn - mean * mean;
  const float rstd = rsqrtf(var + 1e-5f);
#pragma unroll
  for (int j = 0; j < 4; ++j) {
    const int c = tid + j * 256;
    out[(size_t)row * Cn + c] = (bf16)((v4[j] - mean) * rstd * w[c] + b[c]);
  }
}

// ---------------------------------------------------------------------------
// GEMM: out[M,N] = A[M,K](bf16) @ W[N,K]^T(bf16) + bias, with epilogues.
// Each wave: 16 rows x 64 cols (4 16x16 accumulators), K-loop step 32.
// EPI: 0=bf16  1=bf16(exact GELU)  2=f32 resid add  3=bf16 V-transposed  4=f32
// ---------------------------------------------------------------------------
template <int EPI>
__global__ void __launch_bounds__(128) gemm_kernel(
    const bf16* __restrict__ A, const bf16* __restrict__ W,
    const float* __restrict__ bias, const float* __restrict__ resid,
    float* __restrict__ outf, bf16* __restrict__ outb,
    int M, int N, int K, int Tn, int Ctot) {
  const int lane = threadIdx.x & 31;
  const int wave = uni(threadIdx.x >> 5);          // SGPR wave id
  const int n0 = (blockIdx.x * 4 + wave) * 64;
  const int m0 = blockIdx.y * 16;

  v8f acc[4];
#pragma unroll
  for (int t = 0; t < 4; ++t) acc[t] = vzero8();

  const bf16* Ab = A + (size_t)m0 * K;
  const bf16* Wb = W + (size_t)n0 * K;

  for (int k0 = 0; k0 < K; k0 += 32) {
    __builtin_prefetch(Wb + k0 + 512, 0, 1);  // -> global_prefetch_b8
    v16bf af = load_a_frag(Ab + k0, K, lane);
#pragma unroll
    for (int t = 0; t < 4; ++t) {
      v16bf bfg = load_b_frag(Wb + (size_t)(t * 16) * K + k0, K, lane);
      acc[t] = wmma_bf16(af, bfg, acc[t]);
    }
  }

  // C/D layout: acc[t][v] is element (row m0 + v + 8*hi, col n0 + 16*t + nl)
  const int hi = lane >> 4;
  const int nl = lane & 15;
#pragma unroll
  for (int t = 0; t < 4; ++t) {
    const int c = n0 + t * 16 + nl;
    const float bv = (EPI == 4) ? 0.0f : bias[c];
#pragma unroll
    for (int v = 0; v < 8; ++v) {
      const int r = m0 + v + hi * 8;
      float val = acc[t][v] + bv;
      const size_t o = (size_t)r * N + c;
      if (EPI == 0) {
        outb[o] = (bf16)val;
      } else if (EPI == 1) {
        val = 0.5f * val * (1.0f + erff(val * 0.70710678118654752f));
        outb[o] = (bf16)val;
      } else if (EPI == 2) {
        outf[o] = resid[o] + val;
      } else if (EPI == 3) {
        // transposed V store: vT[b][channel c][token] (channel = h*D + d)
        const int bidx = r / Tn;
        const int trow = r - bidx * Tn;
        outb[(size_t)(bidx * Ctot + c) * Tn + trow] = (bf16)val;
      } else {
        outf[o] = val;
      }
    }
  }
}

// ---------------------------------------------------------------------------
// Flash attention, one wave per (b, h, 16-query tile).
// Computes S^T = K @ Q^T so the f32 C/D layout of S^T matches the bf16
// A-fragment layout of P for the P@V WMMA with no cross-lane movement.
// Online softmax; lane<->lane^16 reductions via ds_bpermute.
// All loop bounds are scalar (readfirstlane) -> uniform branches, EXEC
// stays all-1s across every WMMA.
// ---------------------------------------------------------------------------
__global__ void __launch_bounds__(128) attn_kernel(
    const bf16* __restrict__ q, const bf16* __restrict__ k,
    const bf16* __restrict__ vt, bf16* __restrict__ y,
    int Tn, int Hn, int Dn, float scale) {
  const int lane = threadIdx.x & 31;
  const int w = uni(blockIdx.x * (blockDim.x >> 5) + (threadIdx.x >> 5));
  const int ntile = Tn >> 4;
  const int b = w / (Hn * ntile);
  const int rem = w - b * (Hn * ntile);
  const int h = rem / ntile;
  const int i = rem - h * ntile;
  const int C_ = Hn * Dn;
  const int q0 = i * 16;

  const bf16* Qb = q + ((size_t)b * Tn + q0) * C_ + h * Dn;   // stride C_
  const bf16* Kb = k + (size_t)b * Tn * C_ + h * Dn;          // stride C_
  const bf16* Vb = vt + ((size_t)b * C_ + h * Dn) * Tn;       // stride Tn

  // Q^T as B-matrix: B[d][q] = Q[q][d]; two d-steps cover D=64.
  const v16bf bq0 = load_b_frag(Qb, C_, lane);
  const v16bf bq1 = load_b_frag(Qb + 32, C_, lane);

  v8f O[4];
#pragma unroll
  for (int t = 0; t < 4; ++t) O[t] = vzero8();
  float mI = NEGINF, lI = 0.0f;
  const int hi = lane >> 4;
  const int ql = lane & 15;
  const int qg = q0 + ql;   // this lane's global query index

  for (int jb = 0; jb <= q0; jb += 32) {        // 32-key chunks up to diagonal
    v8f st0 = vzero8(), st1 = vzero8();
    {
      v16bf a0 = load_a_frag(Kb + (size_t)jb * C_, C_, lane);
      v16bf a1 = load_a_frag(Kb + (size_t)jb * C_ + 32, C_, lane);
      st0 = wmma_bf16(a0, bq0, st0);
      st0 = wmma_bf16(a1, bq1, st0);
    }
    if (jb + 16 <= q0) {                        // scalar condition
      v16bf a0 = load_a_frag(Kb + (size_t)(jb + 16) * C_, C_, lane);
      v16bf a1 = load_a_frag(Kb + (size_t)(jb + 16) * C_ + 32, C_, lane);
      st1 = wmma_bf16(a0, bq0, st1);
      st1 = wmma_bf16(a1, bq1, st1);
    }
    // S^T layout: st0[v] = S[qg][jb + v + 8*hi]; st1[v] = S[qg][jb+16 + v + 8*hi]
    float s0[8], s1[8];
    float mloc = NEGINF;
#pragma unroll
    for (int v = 0; v < 8; ++v) {
      const int key0 = jb + v + hi * 8;
      const int key1 = key0 + 16;
      s0[v] = (key0 <= qg) ? st0[v] * scale : NEGINF;
      s1[v] = (key1 <= qg) ? st1[v] * scale : NEGINF;
      mloc = fmaxf(mloc, fmaxf(s0[v], s1[v]));
    }
    const float mrow = fmaxf(mloc, lane_xor16(mloc, lane));
    const float mnew = fmaxf(mI, mrow);
    float p0[8], p1[8], ss = 0.0f;
#pragma unroll
    for (int v = 0; v < 8; ++v) {
      p0[v] = __expf(s0[v] - mnew);
      p1[v] = __expf(s1[v] - mnew);
      ss += p0[v] + p1[v];
    }
    const float rowsum = ss + lane_xor16(ss, lane);
    const float fc = __expf(mI - mnew);
    lI = lI * fc + rowsum;
    mI = mnew;
    // rescale O: row r = v + 8*hi needs factor held at lane r
#pragma unroll
    for (int v = 0; v < 8; ++v) {
      const float f = lane_bcast(fc, v + hi * 8);
#pragma unroll
      for (int t = 0; t < 4; ++t) O[t][v] *= f;
    }
    // P as A-fragment: lane layout already correct (see header comment)
    v16bf pa;
#pragma unroll
    for (int v = 0; v < 8; ++v) { pa[v] = (bf16)p0[v]; pa[v + 8] = (bf16)p1[v]; }
#pragma unroll
    for (int t = 0; t < 4; ++t) {
      v16bf bvf = load_b_frag(Vb + (size_t)(t * 16) * Tn + jb, Tn, lane);
      O[t] = wmma_bf16(pa, bvf, O[t]);
    }
  }

  const float linv = 1.0f / lI;
#pragma unroll
  for (int v = 0; v < 8; ++v) {
    const float f = lane_bcast(linv, v + hi * 8);
    const int r = q0 + v + hi * 8;
#pragma unroll
    for (int t = 0; t < 4; ++t) {
      y[((size_t)b * Tn + r) * C_ + h * Dn + t * 16 + ql] = (bf16)(O[t][v] * f);
    }
  }
}

// ---------------------------------------------------------------------------
// Host-side orchestration (all launches on `stream`, graph-capture safe).
// ---------------------------------------------------------------------------
extern "C" void kernel_launch(void* const* d_in, const int* in_sizes, int n_in,
                              void* d_out, int out_size, void* d_ws, size_t ws_size,
                              hipStream_t stream) {
  (void)in_sizes; (void)n_in; (void)out_size; (void)ws_size;
  const int Bn = 2, Tn = 1024, Cn = 1024, Hn = 16, Ln = 6, Vn = 8192, Dn = 64;
  const int Mn = Bn * Tn;  // 2048 tokens

  const int*   idx   = (const int*)  d_in[0];
  const float* tok   = (const float*)d_in[1];
  const float* pos   = (const float*)d_in[2];
  const float* ab    = (const float*)d_in[3];
  const float* lnf_w = (const float*)d_in[4];
  const float* lnf_b = (const float*)d_in[5];
  const float* headw = (const float*)d_in[6];
  const float* ln1w  = (const float*)d_in[7];
  const float* ln1b  = (const float*)d_in[8];
  const float* Wq    = (const float*)d_in[9];
  const float* bq    = (const float*)d_in[10];
  const float* Wk    = (const float*)d_in[11];
  const float* bk    = (const float*)d_in[12];
  const float* Wv    = (const float*)d_in[13];
  const float* bvv   = (const float*)d_in[14];
  const float* Wp    = (const float*)d_in[15];
  const float* bp    = (const float*)d_in[16];
  const float* ln2w  = (const float*)d_in[17];
  const float* ln2b  = (const float*)d_in[18];
  const float* W1    = (const float*)d_in[19];
  const float* b1    = (const float*)d_in[20];
  const float* W2    = (const float*)d_in[21];
  const float* b2    = (const float*)d_in[22];
  float* logits = (float*)d_out;

  // workspace carve (~65 MB)
  char* p = (char*)d_ws;
  auto carve = [&](size_t bytes) {
    void* r = (void*)p;
    p += (bytes + 255) & ~(size_t)255;
    return r;
  };
  float* x  = (float*)carve((size_t)Mn * Cn * 4);        // residual stream f32
  bf16* hb  = (bf16*) carve((size_t)Mn * Cn * 2);        // LN output bf16
  bf16* qb  = (bf16*) carve((size_t)Mn * Cn * 2);
  bf16* kb  = (bf16*) carve((size_t)Mn * Cn * 2);
  bf16* vtb = (bf16*) carve((size_t)Mn * Cn * 2);        // V transposed [B,H,D,T]
  bf16* yb  = (bf16*) carve((size_t)Mn * Cn * 2);        // attention output
  bf16* a1b = (bf16*) carve((size_t)Mn * 4 * Cn * 2);    // MLP hidden
  bf16* wst = (bf16*) carve((size_t)Vn * Cn * 2);        // weight staging (16MB)

  auto cvt = [&](const float* src, bf16* dst, size_t n) {
    cvt_kernel<<<2048, 256, 0, stream>>>(src, dst, (long)n);
  };

  embed_kernel<<<Mn, 256, 0, stream>>>(idx, tok, pos, ab, x, Tn, Cn);

  const size_t wcc = (size_t)Cn * Cn;
  const size_t w4c = (size_t)4 * Cn * Cn;
  const dim3 gCC(Cn / 256, Mn / 16);      // N=1024
  const dim3 g4C(4 * Cn / 256, Mn / 16);  // N=4096
  const dim3 gV(Vn / 256, Mn / 16);       // N=8192

  for (int l = 0; l < Ln; ++l) {
    // --- attention block ---
    ln_kernel<<<Mn, 256, 0, stream>>>(x, ln1w + (size_t)l * Cn, ln1b + (size_t)l * Cn, hb, Cn);
    cvt(Wq + (size_t)l * wcc, wst, wcc);
    gemm_kernel<0><<<gCC, 128, 0, stream>>>(hb, wst, bq + (size_t)l * Cn, nullptr,
                                            nullptr, qb, Mn, Cn, Cn, Tn, Cn);
    cvt(Wk + (size_t)l * wcc, wst, wcc);
    gemm_kernel<0><<<gCC, 128, 0, stream>>>(hb, wst, bk + (size_t)l * Cn, nullptr,
                                            nullptr, kb, Mn, Cn, Cn, Tn, Cn);
    cvt(Wv + (size_t)l * wcc, wst, wcc);
    gemm_kernel<3><<<gCC, 128, 0, stream>>>(hb, wst, bvv + (size_t)l * Cn, nullptr,
                                            nullptr, vtb, Mn, Cn, Cn, Tn, Cn);
    attn_kernel<<<(Bn * Hn * (Tn / 16)) / 4, 128, 0, stream>>>(qb, kb, vtb, yb,
                                                               Tn, Hn, Dn, 0.125f);
    cvt(Wp + (size_t)l * wcc, wst, wcc);
    gemm_kernel<2><<<gCC, 128, 0, stream>>>(yb, wst, bp + (size_t)l * Cn, x,
                                            x, nullptr, Mn, Cn, Cn, Tn, Cn);
    // --- MLP block ---
    ln_kernel<<<Mn, 256, 0, stream>>>(x, ln2w + (size_t)l * Cn, ln2b + (size_t)l * Cn, hb, Cn);
    cvt(W1 + (size_t)l * w4c, wst, w4c);
    gemm_kernel<1><<<g4C, 128, 0, stream>>>(hb, wst, b1 + (size_t)l * 4 * Cn, nullptr,
                                            nullptr, a1b, Mn, 4 * Cn, Cn, Tn, Cn);
    cvt(W2 + (size_t)l * w4c, wst, w4c);
    gemm_kernel<2><<<gCC, 128, 0, stream>>>(a1b, wst, b2 + (size_t)l * Cn, x,
                                            x, nullptr, Mn, Cn, 4 * Cn, Tn, Cn);
  }

  // final LN + vocab head
  ln_kernel<<<Mn, 256, 0, stream>>>(x, lnf_w, lnf_b, hb, Cn);
  cvt(headw, wst, (size_t)Vn * Cn);
  gemm_kernel<4><<<gV, 128, 0, stream>>>(hb, wst, nullptr, nullptr,
                                         logits, nullptr, Mn, Vn, Cn, Tn, Cn);
}
